// CrossAttention_41540923686962
// MI455X (gfx1250) — matrix operand order repaired and checked by reading, not compiled, not run on hardware
//
#include <hip/hip_runtime.h>
#include <hip/hip_bf16.h>

// ---------------------------------------------------------------------------
// Cross-attention for MI455X (gfx1250), wave32 + WMMA f16->f32.
//   B=8, C=256, H=W=48 -> N=2304 tokens.
//   Kernel 0: convert Wq/Wk/Wv f32 -> f16 (one-shot)
//   Kernel 1: Q/K/V projections (f32 acts -> f16 [B,N,C] in workspace)
//   Kernel 2: flash-attention streaming over 32-key blocks, f32 accumulators,
//             register-resident Q fragments + global_prefetch of next block.
// Workspace use: 3*B*N*C f16 (28.3 MB) + 3*C*C f16 (384 KB).
// ---------------------------------------------------------------------------

typedef __attribute__((ext_vector_type(16))) _Float16 v16h;
typedef __attribute__((ext_vector_type(8)))  _Float16 v8h;
typedef __attribute__((ext_vector_type(8)))  float    v8f;

constexpr int BB = 8;
constexpr int CC = 256;
constexpr int NN = 2304;      // 48*48
constexpr int TQ = 64;        // attention query tile per workgroup (4 waves x 16)
constexpr int TK = 32;        // key block per iteration
constexpr int PT = 32;        // projection token tile per workgroup (2 waves x 16)

__device__ __forceinline__ v8f wmma_f16(v16h a, v16h b, v8f c) {
  // D = A(16x32 f16) * B(32x16 f16) + C(16x16 f32)
  return __builtin_amdgcn_wmma_f32_16x16x32_f16(false, a, false, b,
                                                (short)0, c, false, false);
}

// A-fragment (16x32 f16): lane holds row (lane&15); halves 0..7 = K kbase..kbase+7,
// halves 8..15 = K kbase+16..kbase+23, kbase = (lane>>4)*8. Two 16B reads.
__device__ __forceinline__ v16h frag_a(const _Float16* tile, int stride, int koff) {
  const int lane = threadIdx.x & 31;
  const _Float16* p = tile + (lane & 15) * stride + koff + ((lane >> 4) << 3);
  v8h lo = *(const v8h*)p;
  v8h hi = *(const v8h*)(p + 16);
  return __builtin_shufflevector(lo, hi, 0,1,2,3,4,5,6,7,8,9,10,11,12,13,14,15);
}

// B-fragment (32x16 f16), tile stored as [N][K] (column rows of length K):
// lane holds column (lane&15); halves 0..7 = K kbase..kbase+7,
// halves 8..15 = K kbase+16..kbase+23, kbase = koff + (lane>>4)*8.
__device__ __forceinline__ v16h frag_b(const _Float16* tile, int stride, int koff) {
  const int lane = threadIdx.x & 31;
  const _Float16* p = tile + (lane & 15) * stride + koff + ((lane >> 4) << 3);
  v8h lo = *(const v8h*)p;
  v8h hi = *(const v8h*)(p + 16);
  return __builtin_shufflevector(lo, hi, 0,1,2,3,4,5,6,7,8,9,10,11,12,13,14,15);
}

// ---------------------------------------------------------------------------
// Kernel 0: one-shot f32 -> f16 weight conversion (3 matrices of C*C).
// ---------------------------------------------------------------------------
__global__ __launch_bounds__(256)
void wconvert_kernel(const float* __restrict__ Wq, const float* __restrict__ Wk,
                     const float* __restrict__ Wv, _Float16* __restrict__ Wh) {
  const int i = blockIdx.x * blockDim.x + threadIdx.x;   // float4 index, < 49152
  const int m = i / (CC * CC / 4);                        // which matrix
  const int o = i - m * (CC * CC / 4);
  const float* src = (m == 0) ? Wq : (m == 1) ? Wk : Wv;
  float4 v = *(const float4*)(src + o * 4);
  _Float16* d = Wh + (size_t)m * CC * CC + o * 4;
  d[0] = (_Float16)v.x; d[1] = (_Float16)v.y;
  d[2] = (_Float16)v.z; d[3] = (_Float16)v.w;
}

// ---------------------------------------------------------------------------
// Kernel 1: Q = Wq*x + bq (from x), K/V likewise from y; f16 [B,N,C] out.
// grid (NN/PT, BB), block 64 (2 waves, 16 tokens each). Weights are f16.
// ---------------------------------------------------------------------------
__global__ __launch_bounds__(64)
void proj_kernel(const float* __restrict__ x, const float* __restrict__ y,
                 const _Float16* __restrict__ Wh,
                 const float* __restrict__ bq, const float* __restrict__ bk,
                 const float* __restrict__ bv,
                 _Float16* __restrict__ Qh, _Float16* __restrict__ Kh,
                 _Float16* __restrict__ Vh) {
  __shared__ __align__(16) _Float16 Xs[PT * CC];   // token-major [t][c]
  __shared__ __align__(16) _Float16 Ys[PT * CC];

  const int tid  = threadIdx.x;
  const int lane = tid & 31;
  const int wave = tid >> 5;
  const int b    = blockIdx.y;
  const int n0   = blockIdx.x * PT;

  // Stage (transpose [C][N] -> [t][c]), coalesced along tokens.
  for (int i = tid; i < PT * CC; i += 64) {
    const int t = i & (PT - 1);
    const int c = i >> 5;
    const size_t g = ((size_t)(b * CC + c)) * NN + n0 + t;
    Xs[t * CC + c] = (_Float16)x[g];
    Ys[t * CC + c] = (_Float16)y[g];
  }
  __syncthreads();

  const int qr = wave * 16;   // token rows for this wave

  for (int p = 0; p < 3; ++p) {
    const _Float16* src = (p == 0) ? Xs : Ys;
    const _Float16* W   = Wh + (size_t)p * CC * CC;           // [cout][cin] f16
    const float* bias   = (p == 0) ? bq : (p == 1) ? bk : bv;
    _Float16* dst       = (p == 0) ? Qh : (p == 1) ? Kh : Vh;

    // Activation A-fragments are invariant across output tiles: keep in regs.
    v16h a[CC / 32];
#pragma unroll
    for (int kk = 0; kk < CC / 32; ++kk) a[kk] = frag_a(src + qr * CC, CC, kk * 32);

    for (int ct = 0; ct < CC / 16; ++ct) {
      const int cout = ct * 16 + (lane & 15);
      const float bv0 = bias[cout];
      v8f acc;
#pragma unroll
      for (int r = 0; r < 8; ++r) acc[r] = bv0;
#pragma unroll
      for (int kk = 0; kk < CC / 32; ++kk) {
        v16h w = frag_b(W + (size_t)(ct * 16) * CC, CC, kk * 32);
        acc = wmma_f16(a[kk], w, acc);
      }
#pragma unroll
      for (int r = 0; r < 8; ++r) {
        const int m = qr + ((lane >> 4) << 3) + r;
        dst[((size_t)(b * NN + n0 + m)) * CC + cout] = (_Float16)acc[r];
      }
    }
  }
}

// ---------------------------------------------------------------------------
// Kernel 2: flash attention. grid (NN/TQ, BB), block 128 (4 waves x 16 queries).
// LDS: K block 16KB + V^T block 16KB + P strip 4KB = 36KB.
// ---------------------------------------------------------------------------
__global__ __launch_bounds__(128)
void attn_kernel(const _Float16* __restrict__ Qh, const _Float16* __restrict__ Kh,
                 const _Float16* __restrict__ Vh, float* __restrict__ out) {
  __shared__ __align__(16) _Float16 Ks [TK * CC];   // [key][c]
  __shared__ __align__(16) _Float16 Vts[CC * TK];   // [c][key]
  __shared__ __align__(16) _Float16 Ps [TQ * TK];   // [q][key], wave-private strips

  const int tid  = threadIdx.x;
  const int lane = tid & 31;
  const int wave = tid >> 5;
  const int b    = blockIdx.y;
  const int q0   = blockIdx.x * TQ;
  const int qr   = wave * 16;

  // Q tile A-fragments: loop-invariant -> register resident (8 x v16h).
  const _Float16* Qtile = Qh + ((size_t)(b * NN + q0 + qr)) * CC;
  v16h qa[CC / 32];
#pragma unroll
  for (int kk = 0; kk < CC / 32; ++kk) qa[kk] = frag_a(Qtile, CC, kk * 32);

  v8f O[CC / 16];
#pragma unroll
  for (int t = 0; t < CC / 16; ++t)
#pragma unroll
    for (int r = 0; r < 8; ++r) O[t][r] = 0.0f;

  float mi[8], li[8];
#pragma unroll
  for (int r = 0; r < 8; ++r) { mi[r] = -1.0e30f; li[r] = 0.0f; }

  for (int kb = 0; kb < NN / TK; ++kb) {
    const int k0 = kb * TK;
    __syncthreads();   // protect LDS from previous iteration's readers

    // Stage K block row-major (16B chunks, coalesced).
    for (int i = tid; i < (TK * CC) / 8; i += 128) {
      const int k = i >> 5;
      const int c = (i & 31) << 3;
      *(uint4*)&Ks[k * CC + c] =
          *(const uint4*)&Kh[((size_t)(b * NN + k0 + k)) * CC + c];
    }
    // Stage V block transposed to [c][key].
    for (int i = tid; i < (TK * CC) / 8; i += 128) {
      const int k = i >> 5;
      const int c = (i & 31) << 3;
      v8h v = *(const v8h*)&Vh[((size_t)(b * NN + k0 + k)) * CC + c];
#pragma unroll
      for (int j = 0; j < 8; ++j) Vts[(c + j) * TK + k] = v[j];
    }
    __syncthreads();

    // Prefetch next K/V block into cache while this block computes
    // (lowers to global_prefetch_b8; no LOADcnt cost).
    if (kb + 1 < NN / TK) {
      const size_t nb = ((size_t)(b * NN + k0 + TK)) * CC + (size_t)tid * 64;
      __builtin_prefetch(&Kh[nb], 0, 3);
      __builtin_prefetch(&Vh[nb], 0, 3);
    }

    // S = Q * K^T  (16 queries x 32 keys), reduce over C=256.
    v8f S0 = {0.f,0.f,0.f,0.f,0.f,0.f,0.f,0.f};
    v8f S1 = {0.f,0.f,0.f,0.f,0.f,0.f,0.f,0.f};
#pragma unroll
    for (int kk = 0; kk < CC / 32; ++kk) {
      v16h b0 = frag_b(&Ks[0],       CC, kk * 32);
      v16h b1 = frag_b(&Ks[16 * CC], CC, kk * 32);
      S0 = wmma_f16(qa[kk], b0, S0);
      S1 = wmma_f16(qa[kk], b1, S1);
    }

    // Online softmax: per-row max/sum across 16 lanes sharing a row.
#pragma unroll
    for (int r = 0; r < 8; ++r) {
      float rm = fmaxf(S0[r], S1[r]);
      rm = fmaxf(rm, __shfl_xor(rm, 1));
      rm = fmaxf(rm, __shfl_xor(rm, 2));
      rm = fmaxf(rm, __shfl_xor(rm, 4));
      rm = fmaxf(rm, __shfl_xor(rm, 8));
      const float mn = fmaxf(mi[r], rm);
      const float sc = __expf(mi[r] - mn);
      const float p0 = __expf(S0[r] - mn);
      const float p1 = __expf(S1[r] - mn);
      const int row = qr + ((lane >> 4) << 3) + r;
      Ps[row * TK +      (lane & 15)] = (_Float16)p0;
      Ps[row * TK + 16 + (lane & 15)] = (_Float16)p1;
      float rs = p0 + p1;
      rs += __shfl_xor(rs, 1);
      rs += __shfl_xor(rs, 2);
      rs += __shfl_xor(rs, 4);
      rs += __shfl_xor(rs, 8);
      li[r] = li[r] * sc + rs;
      mi[r] = mn;
#pragma unroll
      for (int t = 0; t < CC / 16; ++t) O[t][r] *= sc;
    }

    // O += P * V   (A = wave-private P strip, B = V^T columns).
    v16h pa = frag_a(&Ps[qr * TK], TK, 0);
#pragma unroll
    for (int t = 0; t < CC / 16; ++t) {
      v16h bv = frag_b(&Vts[(t * 16) * TK], TK, 0);
      O[t] = wmma_f16(pa, bv, O[t]);
    }
  }

  // Normalize and store to [B][C][N] layout.
  float inv[8];
#pragma unroll
  for (int r = 0; r < 8; ++r) inv[r] = 1.0f / li[r];
#pragma unroll
  for (int t = 0; t < CC / 16; ++t) {
    const int c = t * 16 + (lane & 15);
    float* op = out + ((size_t)(b * CC + c)) * NN + q0 + qr + ((lane >> 4) << 3);
#pragma unroll
    for (int r = 0; r < 8; ++r) op[r] = O[t][r] * inv[r];
  }
}

// ---------------------------------------------------------------------------
extern "C" void kernel_launch(void* const* d_in, const int* in_sizes, int n_in,
                              void* d_out, int out_size, void* d_ws, size_t ws_size,
                              hipStream_t stream) {
  const float* x  = (const float*)d_in[0];
  const float* y  = (const float*)d_in[1];
  const float* Wq = (const float*)d_in[2];
  const float* bq = (const float*)d_in[3];
  const float* Wk = (const float*)d_in[4];
  const float* bk = (const float*)d_in[5];
  const float* Wv = (const float*)d_in[6];
  const float* bv = (const float*)d_in[7];
  float* out = (float*)d_out;

  const size_t elems = (size_t)BB * NN * CC;   // 4,718,592 per tensor
  const size_t wele  = (size_t)3 * CC * CC;    // 196,608 weight halves
  if (ws_size < (3 * elems + wele) * sizeof(_Float16)) return;  // ~28.7 MB
  _Float16* Qh = (_Float16*)d_ws;
  _Float16* Kh = Qh + elems;
  _Float16* Vh = Kh + elems;
  _Float16* Wh = Vh + elems;

  wconvert_kernel<<<dim3((3 * CC * CC / 4) / 256), 256, 0, stream>>>(Wq, Wk, Wv, Wh);
  proj_kernel<<<dim3(NN / PT, BB), 64, 0, stream>>>(x, y, Wh, bq, bk, bv,
                                                    Qh, Kh, Vh);
  attn_kernel<<<dim3(NN / TQ, BB), 128, 0, stream>>>(Qh, Kh, Vh, out);
}
